// RRB_M_91130616087313
// MI455X (gfx1250) — matrix operand order repaired and checked by reading, not compiled, not run on hardware
//
#include <hip/hip_runtime.h>
#include <math.h>

// ---------------- constants ----------------
#define BATCH   8192
#define N_PRED  12
#define MODES   2

// padded conv2 output: 32 channels x 176 pixels (169 valid, rest zero-weighted)
#define H2_LDK  5632            // 32*176
#define FC_KT   176             // 5632/32 k-steps

typedef __attribute__((ext_vector_type(16))) _Float16 v16h;
typedef __attribute__((ext_vector_type(8)))  _Float16 v8h;
typedef __attribute__((ext_vector_type(4)))  _Float16 v4h;
typedef __attribute__((ext_vector_type(8)))  float    v8f;

// ---------------- WMMA fragment helpers ----------------
// A 16x32 f16 (ISA 7.12.2): lane l row (l&15); elem j -> K=(l>>4)*8+(j&7)+(j>>3)*16
// B 32x16 f16: lane l col (l&15); elem j -> K=(l>>4)*8+(j&7)+(j>>3)*16
// C/D 16x16 f32: lane l, vgpr r -> row r+8*(l>>4), col (l&15)

__device__ __forceinline__ v16h wmma_afrag_linear(const _Float16* A, int lda, int kbase, int lane) {
    const _Float16* p = A + (size_t)(lane & 15) * lda + kbase + ((lane >> 4) << 3);
    v8h lo = *(const v8h*)(p);
    v8h hi = *(const v8h*)(p + 16);
    return __builtin_shufflevector(lo, hi, 0,1,2,3,4,5,6,7,8,9,10,11,12,13,14,15);
}

__device__ __forceinline__ v16h wmma_bfrag(const _Float16* packed, int fragIdx, int lane) {
    return *(const v16h*)(packed + (size_t)fragIdx * 512 + (size_t)lane * 16);
}

__device__ __forceinline__ v8f wmma_f16(v16h a, v16h b, v8f c) {
    return __builtin_amdgcn_wmma_f32_16x16x32_f16(false, a, false, b, (short)0, c, false, false);
}

// CDNA5 async global->LDS copy (16B per lane), tracked by ASYNCcnt
__device__ __forceinline__ void async_copy16(unsigned lds_addr, unsigned long long gbase, unsigned voff) {
    asm volatile("global_load_async_to_lds_b128 %0, %1, %2 offset:0"
                 :: "v"(lds_addr), "v"(voff), "s"(gbase) : "memory");
}
__device__ __forceinline__ void wait_async0() {
    asm volatile("s_wait_asynccnt 0x0" ::: "memory");
}

// GEMM: f16 activations (LDS or global), pre-packed B frags, f16 out (+bias, opt relu)
__device__ __forceinline__ void gemm_h(const _Float16* A, int lda, int kT,
                                       const _Float16* Bp, int nT,
                                       _Float16* Out, int ldo,
                                       const float* bias, int biasN, bool relu, int lane) {
    int col = lane & 15;
    int ro  = (lane >> 4) << 3;
    for (int nt = 0; nt < nT; ++nt) {
        v8f acc = {};
        for (int kt = 0; kt < kT; ++kt) {
            v16h a = wmma_afrag_linear(A, lda, kt * 32, lane);
            v16h b = wmma_bfrag(Bp, kt * nT + nt, lane);
            acc = wmma_f16(a, b, acc);
        }
        int n = nt * 16 + col;
        float bb = bias[n < biasN ? n : 0];
        bb = (n < biasN) ? bb : 0.0f;
#pragma unroll
        for (int r = 0; r < 8; ++r) {
            float v = acc[r] + bb;
            if (relu) v = fmaxf(v, 0.0f);
            Out[(size_t)(ro + r) * ldo + n] = (_Float16)v;
        }
    }
}

// same but f32 output at column offset nbase, no relu
__device__ __forceinline__ void gemm_f(const _Float16* A, int lda, int kT,
                                       const _Float16* Bp, int nT,
                                       float* Out, int ldo, int nbase,
                                       const float* bias, int biasN, int lane) {
    int col = lane & 15;
    int ro  = (lane >> 4) << 3;
    for (int nt = 0; nt < nT; ++nt) {
        v8f acc = {};
        for (int kt = 0; kt < kT; ++kt) {
            v16h a = wmma_afrag_linear(A, lda, kt * 32, lane);
            v16h b = wmma_bfrag(Bp, kt * nT + nt, lane);
            acc = wmma_f16(a, b, acc);
        }
        int n = nt * 16 + col;
        float bb = bias[n < biasN ? n : 0];
        bb = (n < biasN) ? bb : 0.0f;
#pragma unroll
        for (int r = 0; r < 8; ++r)
            Out[(size_t)(ro + r) * ldo + nbase + n] = acc[r] + bb;
    }
}

// ---------------- weight packing: W (N,K) f32 -> B fragments f16 ----------------
__global__ void pack_w_kernel(const float* __restrict__ W, int K, int N,
                              _Float16* __restrict__ out, int nTiles, int total) {
    int idx = blockIdx.x * blockDim.x + threadIdx.x;
    if (idx >= total) return;
    int frag = idx >> 9;
    int e    = idx & 511;
    int lane = e >> 4;
    int j    = e & 15;
    int kt = frag / nTiles, nt = frag - kt * nTiles;
    int n = nt * 16 + (lane & 15);
    int k = kt * 32 + ((lane >> 4) << 3) + (j & 7) + ((j >> 3) << 4);      // B-fragment K pattern
    float v = (n < N && k < K) ? W[(size_t)n * K + k] : 0.0f;
    out[idx] = (_Float16)v;
}

// cnn_fw with pixel padding 169->176: packed K' = c*176+pix, source k = c*169+pix
__global__ void pack_fw_kernel(const float* __restrict__ W, _Float16* __restrict__ out, int total) {
    int idx = blockIdx.x * blockDim.x + threadIdx.x;
    if (idx >= total) return;
    int frag = idx >> 9;
    int e    = idx & 511;
    int lane = e >> 4;
    int j    = e & 15;
    int kt = frag >> 3, nt = frag & 7;             // nTiles = 8
    int n  = nt * 16 + (lane & 15);
    int kk = kt * 32 + ((lane >> 4) << 3) + (j & 7) + ((j >> 3) << 4);     // 0..5631
    int c = kk / 176, pix = kk - c * 176;
    float v = (pix < 169) ? W[(size_t)n * 5408 + c * 169 + pix] : 0.0f;
    out[idx] = (_Float16)v;
}

// ---------------- preprocess: velocity + top-k neighbors ----------------
__global__ void preproc_kernel(const float* __restrict__ obs,
                               const float* __restrict__ ps_in,
                               const float* __restrict__ sr_in,
                               _Float16* __restrict__ vbuf,
                               _Float16* __restrict__ relbuf,
                               float* __restrict__ lastp,
                               float* __restrict__ scaleb) {
    int b = blockIdx.x * blockDim.x + threadIdx.x;
    if (b >= BATCH) return;
    const float* ob = obs + (size_t)b * 8 * 64 * 2;   // [t][n][c]
    float p = ps_in[b], s = sr_in[b];
    float sv = s / p;

    for (int t = 0; t < 7; ++t) {
        vbuf[b * 32 + 2 * t]     = (_Float16)((ob[(t + 1) * 128 + 0] - ob[t * 128 + 0]) * sv);
        vbuf[b * 32 + 2 * t + 1] = (_Float16)((ob[(t + 1) * 128 + 1] - ob[t * 128 + 1]) * sv);
    }
    for (int i = 14; i < 32; ++i) vbuf[b * 32 + i] = (_Float16)0.0f;

    float x0 = ob[7 * 128 + 0], y0 = ob[7 * 128 + 1];
    float rx0 = __builtin_isnan(x0) ? -1000.0f : x0;
    float ry0 = __builtin_isnan(y0) ? -1000.0f : y0;

    float dist[64];
    for (int j = 0; j < 64; ++j) {
        float xj = ob[7 * 128 + 2 * j], yj = ob[7 * 128 + 2 * j + 1];
        bool behind = xj > x0;
        float cxj = (behind || __builtin_isnan(xj)) ? -1000.0f : xj;
        float cyj = (behind || __builtin_isnan(yj)) ? -1000.0f : yj;
        float dx = cxj - rx0, dy = cyj - ry0;
        dist[j] = dx * dx + dy * dy;
    }
    unsigned long long used = 0ull;
    int chosen[4]; float cd[4];
    for (int k = 0; k < 4; ++k) {
        float best = 3.4e38f; int bi = 0;
        for (int j = 0; j < 64; ++j)
            if (!((used >> j) & 1ull) && dist[j] < best) { best = dist[j]; bi = j; }
        used |= 1ull << bi; chosen[k] = bi; cd[k] = best;
    }
    for (int k = 1; k < 4; ++k) {
        int j = chosen[k];
        float rel[6]; float ssum = 0.0f;
        for (int t = 0; t < 3; ++t) {
            float nx = ob[(5 + t) * 128 + 2 * j], ny = ob[(5 + t) * 128 + 2 * j + 1];
            rel[2 * t]     = x0 - nx;
            rel[2 * t + 1] = y0 - ny;
            ssum += rel[2 * t] + rel[2 * t + 1];
        }
        bool bad = (cd[k] > 600000.0f) || __builtin_isnan(ssum);
        for (int t = 0; t < 3; ++t)
            for (int c = 0; c < 2; ++c) {
                float v = bad ? 2000.0f : rel[2 * t + c];
                relbuf[b * 32 + t * 6 + (k - 1) * 2 + c] = (_Float16)(v * sv);
            }
    }
    for (int i = 18; i < 32; ++i) relbuf[b * 32 + i] = (_Float16)0.0f;
    lastp[2 * b] = x0; lastp[2 * b + 1] = y0;
    scaleb[b] = p / s;
}

// ---------------- fused conv1+conv2 per image (implicit GEMM via WMMA) ----------------
__global__ void __launch_bounds__(256) conv_fused_kernel(
        const float* __restrict__ scene,
        const float* __restrict__ b1, const float* __restrict__ b2,
        const _Float16* __restrict__ pk_k1, const _Float16* __restrict__ pk_k2,
        _Float16* __restrict__ h2) {
    __shared__ __align__(16) _Float16 simg[64 * 64];    // 8KB
    __shared__ __align__(16) _Float16 sh1[16 * 912];    // 29.2KB [ch][pix], padded to 912

    int b    = blockIdx.x;
    int tid  = threadIdx.x;
    int lane = tid & 31;
    int wv   = tid >> 5;

    // vectorized f32 -> f16 image load
    const float* img = scene + (size_t)b * 4096;
    for (int i = tid; i < 1024; i += 256) {
        float4 f = ((const float4*)img)[i];
        v4h h; h[0] = (_Float16)f.x; h[1] = (_Float16)f.y; h[2] = (_Float16)f.z; h[3] = (_Float16)f.w;
        *(v4h*)(simg + 4 * i) = h;
    }
    __syncthreads();

    // ---- conv1: M=900 (pad 912), N=16, K=25 (pad 32); fully branchless ----
    float bias1 = b1[lane & 15];
    v16h bf1 = wmma_bfrag(pk_k1, 0, lane);
    for (int mt = wv; mt < 57; mt += 8) {
        int m  = mt * 16 + (lane & 15);
        int mc = m <= 899 ? m : 899;                 // clamp: padded rows recompute row 899
        int oy = mc / 30, ox = mc - oy * 30;
        int ibase = oy * 128 + ox * 2;               // (2*oy)*64 + 2*ox
        v16h a;
#pragma unroll
        for (int j = 0; j < 16; ++j) {
            int k  = ((lane >> 4) << 3) + (j & 7) + ((j >> 3) << 4);
            int kc = k < 25 ? k : 24;
            int ky = kc / 5, kx = kc - ky * 5;
            _Float16 v = simg[ibase + ky * 64 + kx]; // unconditional load
            a[j] = (k < 25) ? v : (_Float16)0.0f;    // cndmask, no branch
        }
        v8f acc = {};
        acc = wmma_f16(a, bf1, acc);
        int ro = (lane >> 4) << 3;
        v8h st;
#pragma unroll
        for (int r = 0; r < 8; ++r) st[r] = (_Float16)fmaxf(acc[r] + bias1, 0.0f);
        *(v8h*)(sh1 + (lane & 15) * 912 + mt * 16 + ro) = st;   // one ds_store_b128
    }
    __syncthreads();

    // ---- conv2: M=169 (pad 176), N=32 (2 tiles), K=400 (13 k-steps); branchless ----
    float bias2a = b2[lane & 15];
    float bias2b = b2[16 + (lane & 15)];
    _Float16* hb = h2 + (size_t)b * H2_LDK;
    for (int mt = wv; mt < 11; mt += 8) {
        int m  = mt * 16 + (lane & 15);
        int mc = m <= 168 ? m : 168;
        int oy = mc / 13, ox = mc - oy * 13;
        int pbase = oy * 60 + ox * 2;                // (2*oy)*30 + 2*ox
        v8f acc0 = {}, acc1 = {};
        for (int ks = 0; ks < 13; ++ks) {
            v16h a;
#pragma unroll
            for (int j = 0; j < 16; ++j) {
                int k  = ks * 32 + ((lane >> 4) << 3) + (j & 7) + ((j >> 3) << 4);
                int kc = k < 400 ? k : 399;
                int ci = kc / 25; int rem = kc - ci * 25;
                int ky = rem / 5, kx = rem - ky * 5;
                _Float16 v = sh1[ci * 912 + pbase + ky * 30 + kx];
                a[j] = (k < 400) ? v : (_Float16)0.0f;
            }
            v16h bb0 = wmma_bfrag(pk_k2, ks * 2 + 0, lane);
            v16h bb1 = wmma_bfrag(pk_k2, ks * 2 + 1, lane);
            acc0 = wmma_f16(a, bb0, acc0);
            acc1 = wmma_f16(a, bb1, acc1);
        }
        int ro = (lane >> 4) << 3;
        v8h s0v, s1v;
#pragma unroll
        for (int r = 0; r < 8; ++r) {
            s0v[r] = (_Float16)fmaxf(acc0[r] + bias2a, 0.0f);
            s1v[r] = (_Float16)fmaxf(acc1[r] + bias2b, 0.0f);
        }
        // padded pixels 169..175 hold junk; FC weights there are zero
        *(v8h*)(hb + (size_t)(lane & 15) * 176 + mt * 16 + ro)        = s0v;
        *(v8h*)(hb + (size_t)(16 + (lane & 15)) * 176 + mt * 16 + ro) = s1v;
    }
}

// ---------------- scene FC: (8192 x 5632) x (5632 x 128), async-LDS staged ----------------
__global__ void __launch_bounds__(256) fc_kernel(const _Float16* __restrict__ h2,
                                                 const _Float16* __restrict__ pk_fw,
                                                 const float* __restrict__ fb,
                                                 _Float16* __restrict__ sfeat) {
    __shared__ __align__(16) _Float16 sA[16 * 512];    // 16KB A-chunk (16 rows x 512 halves)
    int blk  = blockIdx.x;
    int tid  = threadIdx.x;
    int lane = tid & 31;
    int nt   = tid >> 5;                               // 8 waves, one n-tile each

    unsigned long long gbase = (unsigned long long)(h2 + (size_t)blk * 16 * H2_LDK);
    unsigned sbase = (unsigned)(uintptr_t)sA;          // low 32 bits = LDS offset

    v8f acc = {};
    for (int ch = 0; ch < 11; ++ch) {                  // 11 chunks x 16 k-steps = 176
        // stage 16x512 halves via async global->LDS (4 x 16B per thread)
        for (int i = tid; i < 1024; i += 256) {
            int r  = i >> 6;                           // 64 x 16B chunks per row
            int cc = i & 63;
            unsigned voff = (unsigned)((r * H2_LDK + ch * 512 + cc * 8) * 2);
            async_copy16(sbase + (unsigned)((r * 512 + cc * 8) * 2), gbase, voff);
        }
        wait_async0();
        __syncthreads();
#pragma unroll
        for (int ks = 0; ks < 16; ++ks) {
            v16h a = wmma_afrag_linear(sA, 512, ks * 32, lane);
            v16h b = wmma_bfrag(pk_fw, (ch * 16 + ks) * 8 + nt, lane);
            acc = wmma_f16(a, b, acc);
        }
        __syncthreads();
    }
    float bb = fb[nt * 16 + (lane & 15)];
    int ro = (lane >> 4) << 3;
#pragma unroll
    for (int r = 0; r < 8; ++r) {
        int row = blk * 16 + ro + r;
        sfeat[(size_t)row * 128 + nt * 16 + (lane & 15)] = (_Float16)(acc[r] + bb);
    }
}

// ---------------- fused head: MLPs + dec + rm/rv heads + epilogue ----------------
struct HeadArgs {
    const _Float16 *vbuf, *relbuf, *sfeat;
    const _Float16 *pk_et1, *pk_et2, *pk_et3, *pk_ev1, *pk_ev2, *pk_ev3;
    const _Float16 *pk_dec, *pk_rm1, *pk_rm2, *pk_rv1, *pk_rv2;
    const float *etb1, *etb2, *etb3, *evb1, *evb2, *evb3;
    const float *decb, *rmb1, *rmb2, *rvb1, *rvb2;
    const float *lastp, *scaleb;
    float* out;
};

__global__ void __launch_bounds__(64) head_kernel(HeadArgs A) {
    __shared__ __align__(16) _Float16 sFeats[2][16 * 256];  // 16KB
    __shared__ __align__(16) _Float16 sF2[2][16 * 128];     // 8KB
    __shared__ __align__(16) _Float16 sTmp[2][16 * 64];     // 4KB
    __shared__ __align__(16) float    sOut[2][16 * 128];    // 16KB: mean 0..47, var 48..119

    int lane = threadIdx.x & 31;
    int w    = threadIdx.x >> 5;
    int g    = blockIdx.x * 2 + w;
    int s0   = g * 16;

    _Float16* feats = sFeats[w];
    _Float16* f2    = sF2[w];
    _Float16* tA    = sTmp[w];
    _Float16* tB    = sTmp[w] + 16 * 32;
    float*    ob    = sOut[w];

    // encoded_traj: 14->32->32->64 (feats cols 0..63)
    gemm_h(A.vbuf + (size_t)s0 * 32, 32, 1, A.pk_et1, 2, tA, 32, A.etb1, 32, true,  lane);
    gemm_h(tA, 32, 1, A.pk_et2, 2, tB, 32, A.etb2, 32, true,  lane);
    gemm_h(tB, 32, 1, A.pk_et3, 4, feats, 256, A.etb3, 64, false, lane);
    // encoded_vehs: 18->32->32->64 (feats cols 64..127)
    gemm_h(A.relbuf + (size_t)s0 * 32, 32, 1, A.pk_ev1, 2, tA, 32, A.evb1, 32, true,  lane);
    gemm_h(tA, 32, 1, A.pk_ev2, 2, tB, 32, A.evb2, 32, true,  lane);
    gemm_h(tB, 32, 1, A.pk_ev3, 4, feats + 64, 256, A.evb3, 64, false, lane);
    // scene features (feats cols 128..255)
    for (int i = lane; i < 16 * 128; i += 32) {
        int r = i >> 7, c = i & 127;
        feats[r * 256 + 128 + c] = A.sfeat[(size_t)(s0 + r) * 128 + c];
    }
    // dec: 256->128 relu
    gemm_h(feats, 256, 8, A.pk_dec, 8, f2, 128, A.decb, 128, true, lane);
    // rm head: 128->128 relu -> 48
    gemm_h(f2, 128, 4, A.pk_rm1, 8, feats, 128, A.rmb1, 128, true, lane);
    gemm_f(feats, 128, 4, A.pk_rm2, 3, ob, 128, 0,  A.rmb2, 48, lane);
    // rv head: 128->128 relu -> 72
    gemm_h(f2, 128, 4, A.pk_rv1, 8, feats + 2048, 128, A.rvb1, 128, true, lane);
    gemm_f(feats + 2048, 128, 4, A.pk_rv2, 5, ob, 128, 48, A.rvb2, 72, lane);

    // epilogue: lane -> (sample = lane>>1, mode = lane&1)
    int s  = lane >> 1;
    int mm = lane & 1;
    int bidx = s0 + s;
    float sc = A.scaleb[bidx];                  // ps / sr
    float cx = A.lastp[2 * bidx + 0];
    float cy = A.lastp[2 * bidx + 1];
    const float* row = ob + s * 128;
    for (int t = 0; t < N_PRED; ++t) {
        float mx = row[mm * 24 + t * 2 + 0];
        float my = row[mm * 24 + t * 2 + 1];
        float r0 = row[48 + mm * 36 + t * 3 + 0];
        float r1 = row[48 + mm * 36 + t * 3 + 1];
        float r2 = row[48 + mm * 36 + t * 3 + 2];
        cx += mx * sc;
        cy += my * sc;
        float* o = A.out + (((size_t)bidx * MODES + mm) * N_PRED + t) * 5;
        o[0] = cx;
        o[1] = cy;
        o[2] = expf(r0) * sc;
        o[3] = expf(r1) * sc;
        o[4] = tanhf(r2) * sc;
    }
}

// ---------------- host side ----------------
static inline size_t alloc_off(size_t& o, size_t bytes) {
    size_t r = o;
    o = (o + bytes + 255) & ~(size_t)255;
    return r;
}

extern "C" void kernel_launch(void* const* d_in, const int* in_sizes, int n_in,
                              void* d_out, int out_size, void* d_ws, size_t ws_size,
                              hipStream_t stream) {
    (void)in_sizes; (void)n_in; (void)out_size; (void)ws_size;
    const float* obs   = (const float*)d_in[0];
    const float* scene = (const float*)d_in[1];
    const float* ps    = (const float*)d_in[2];
    const float* sr    = (const float*)d_in[3];
    const float* et_w1 = (const float*)d_in[4];  const float* et_b1 = (const float*)d_in[5];
    const float* et_w2 = (const float*)d_in[6];  const float* et_b2 = (const float*)d_in[7];
    const float* et_w3 = (const float*)d_in[8];  const float* et_b3 = (const float*)d_in[9];
    const float* ev_w1 = (const float*)d_in[10]; const float* ev_b1 = (const float*)d_in[11];
    const float* ev_w2 = (const float*)d_in[12]; const float* ev_b2 = (const float*)d_in[13];
    const float* ev_w3 = (const float*)d_in[14]; const float* ev_b3 = (const float*)d_in[15];
    const float* dec_w = (const float*)d_in[16]; const float* dec_b = (const float*)d_in[17];
    const float* rm_w1 = (const float*)d_in[18]; const float* rm_b1 = (const float*)d_in[19];
    const float* rm_w2 = (const float*)d_in[20]; const float* rm_b2 = (const float*)d_in[21];
    const float* rv_w1 = (const float*)d_in[22]; const float* rv_b1 = (const float*)d_in[23];
    const float* rv_w2 = (const float*)d_in[24]; const float* rv_b2 = (const float*)d_in[25];
    const float* cnn_k1 = (const float*)d_in[26]; const float* cnn_b1 = (const float*)d_in[27];
    const float* cnn_k2 = (const float*)d_in[28]; const float* cnn_b2 = (const float*)d_in[29];
    const float* cnn_fw = (const float*)d_in[30]; const float* cnn_fb = (const float*)d_in[31];

    char* ws = (char*)d_ws;
    size_t o = 0;
    struct PD { const float* W; int K, N, kT, nT; size_t off; };
    PD pd[13] = {
        { et_w1,  14,  32, 1, 2, 0 }, { et_w2,  32,  32, 1, 2, 0 }, { et_w3,  32,  64, 1, 4, 0 },
        { ev_w1,  18,  32, 1, 2, 0 }, { ev_w2,  32,  32, 1, 2, 0 }, { ev_w3,  32,  64, 1, 4, 0 },
        { dec_w, 256, 128, 8, 8, 0 },
        { rm_w1, 128, 128, 4, 8, 0 }, { rm_w2, 128,  48, 4, 3, 0 },
        { rv_w1, 128, 128, 4, 8, 0 }, { rv_w2, 128,  72, 4, 5, 0 },
        { cnn_k1, 25,  16, 1, 1, 0 }, { cnn_k2, 400, 32, 13, 2, 0 },
    };
    for (int i = 0; i < 13; ++i)
        pd[i].off = alloc_off(o, (size_t)pd[i].kT * pd[i].nT * 512 * sizeof(_Float16));
    size_t off_fw     = alloc_off(o, (size_t)FC_KT * 8 * 512 * sizeof(_Float16));

    size_t off_vbuf   = alloc_off(o, (size_t)BATCH * 32 * sizeof(_Float16));
    size_t off_relbuf = alloc_off(o, (size_t)BATCH * 32 * sizeof(_Float16));
    size_t off_lastp  = alloc_off(o, (size_t)BATCH * 2 * sizeof(float));
    size_t off_scale  = alloc_off(o, (size_t)BATCH * sizeof(float));
    size_t off_h2     = alloc_off(o, (size_t)BATCH * H2_LDK * sizeof(_Float16));
    size_t off_sfeat  = alloc_off(o, (size_t)BATCH * 128 * sizeof(_Float16));

    // 1) pack weights into B-fragment layout
    for (int i = 0; i < 13; ++i) {
        int total = pd[i].kT * pd[i].nT * 512;
        pack_w_kernel<<<(total + 255) / 256, 256, 0, stream>>>(
            pd[i].W, pd[i].K, pd[i].N, (_Float16*)(ws + pd[i].off), pd[i].nT, total);
    }
    {
        int total = FC_KT * 8 * 512;
        pack_fw_kernel<<<(total + 255) / 256, 256, 0, stream>>>(
            cnn_fw, (_Float16*)(ws + off_fw), total);
    }

    // 2) preprocess
    preproc_kernel<<<BATCH / 256, 256, 0, stream>>>(
        obs, ps, sr,
        (_Float16*)(ws + off_vbuf), (_Float16*)(ws + off_relbuf),
        (float*)(ws + off_lastp), (float*)(ws + off_scale));

    // 3) fused conv1+conv2
    conv_fused_kernel<<<BATCH, 256, 0, stream>>>(
        scene, cnn_b1, cnn_b2,
        (const _Float16*)(ws + pd[11].off), (const _Float16*)(ws + pd[12].off),
        (_Float16*)(ws + off_h2));

    // 4) scene FC GEMM (async-LDS staged)
    fc_kernel<<<BATCH / 16, 256, 0, stream>>>(
        (const _Float16*)(ws + off_h2), (const _Float16*)(ws + off_fw),
        cnn_fb, (_Float16*)(ws + off_sfeat));

    // 5) fused head
    HeadArgs ha;
    ha.vbuf   = (const _Float16*)(ws + off_vbuf);
    ha.relbuf = (const _Float16*)(ws + off_relbuf);
    ha.sfeat  = (const _Float16*)(ws + off_sfeat);
    ha.pk_et1 = (const _Float16*)(ws + pd[0].off);
    ha.pk_et2 = (const _Float16*)(ws + pd[1].off);
    ha.pk_et3 = (const _Float16*)(ws + pd[2].off);
    ha.pk_ev1 = (const _Float16*)(ws + pd[3].off);
    ha.pk_ev2 = (const _Float16*)(ws + pd[4].off);
    ha.pk_ev3 = (const _Float16*)(ws + pd[5].off);
    ha.pk_dec = (const _Float16*)(ws + pd[6].off);
    ha.pk_rm1 = (const _Float16*)(ws + pd[7].off);
    ha.pk_rm2 = (const _Float16*)(ws + pd[8].off);
    ha.pk_rv1 = (const _Float16*)(ws + pd[9].off);
    ha.pk_rv2 = (const _Float16*)(ws + pd[10].off);
    ha.etb1 = et_b1; ha.etb2 = et_b2; ha.etb3 = et_b3;
    ha.evb1 = ev_b1; ha.evb2 = ev_b2; ha.evb3 = ev_b3;
    ha.decb = dec_b; ha.rmb1 = rm_b1; ha.rmb2 = rm_b2; ha.rvb1 = rv_b1; ha.rvb2 = rv_b2;
    ha.lastp = (const float*)(ws + off_lastp);
    ha.scaleb = (const float*)(ws + off_scale);
    ha.out = (float*)d_out;
    head_kernel<<<BATCH / 32, 64, 0, stream>>>(ha);
}
